// Sae_26379689132845
// MI455X (gfx1250) — matrix operand order repaired and checked by reading, not compiled, not run on hardware
//
#include <hip/hip_runtime.h>
#include <hip/hip_bf16.h>

typedef __attribute__((ext_vector_type(16))) __bf16 v16bf;
typedef __attribute__((ext_vector_type(8)))  float  v8f;

#define DIN     768
#define NLAT    24576
#define KTOP    64
#define BATCH   8192

#define ROWS_WG 32      // rows per workgroup (two 16-row WMMA tiles)
#define NCHUNK  256     // 8 waves * 32 columns per super-chunk
#define A_STR   776     // 768 + 8 bf16 pad -> conflict-free b128 LDS reads
#define P_STR   260     // 256 + 4 f32 pad (rows r, r+8 land on disjoint banks)

__device__ __forceinline__ int lane_id() {
    return __builtin_amdgcn_mbcnt_lo(~0u, 0);
}
__device__ __forceinline__ float bcastf(float v, int src) {
    return __builtin_bit_cast(float,
        __builtin_amdgcn_ds_bpermute(src << 2, __builtin_bit_cast(int, v)));
}
__device__ __forceinline__ int bcasti(int v, int src) {
    return __builtin_amdgcn_ds_bpermute(src << 2, v);
}

// ---- one-shot W_enc f32 -> bf16 (lives in d_ws, L2-resident thereafter) ----
__global__ __launch_bounds__(256) void wenc_to_bf16(const float* __restrict__ w,
                                                    __bf16* __restrict__ o) {
    size_t i = ((size_t)blockIdx.x * 256 + threadIdx.x) * 8;
    float4 f0 = *(const float4*)(w + i);
    float4 f1 = *(const float4*)(w + i + 4);
    union { __bf16 b[8]; uint4 u; } p;
    p.b[0] = (__bf16)f0.x; p.b[1] = (__bf16)f0.y;
    p.b[2] = (__bf16)f0.z; p.b[3] = (__bf16)f0.w;
    p.b[4] = (__bf16)f1.x; p.b[5] = (__bf16)f1.y;
    p.b[6] = (__bf16)f1.z; p.b[7] = (__bf16)f1.w;
    *(uint4*)(o + i) = p.u;
}

// ---- fused encoder GEMM (bf16 WMMA) + ReLU + top-64 + sparse decode ----
__global__ __launch_bounds__(256) void sae_fused(
    const float* __restrict__ x, const __bf16* __restrict__ wencB,
    const float* __restrict__ b_enc, const float* __restrict__ w_dec,
    const float* __restrict__ b_dec, float* __restrict__ out)
{
    __shared__ __align__(16) __bf16 xcb[ROWS_WG * A_STR];   // (x - b_dec) bf16 tile
    __shared__ float pre_s[ROWS_WG * P_STR];                // relu'd pre-acts chunk

    const int tid  = threadIdx.x;
    const int wave = tid >> 5;
    const int lane = lane_id();
    const int r0   = blockIdx.x * ROWS_WG;
    const int g    = lane >> 4;      // half-wave (0/1)
    const int nl   = lane & 15;      // column within 16-wide tile / A row

    // ---- stage A tile: xc = x - b_dec, as bf16 in LDS ----
    for (int i = tid; i < ROWS_WG * DIN; i += 256) {
        int row = i / DIN, d = i - row * DIN;
        float v = x[(size_t)(r0 + row) * DIN + d] - b_dec[d];
        xcb[row * A_STR + d] = (__bf16)v;
    }
    __syncthreads();

    // ---- per-wave top-64 state for 4 rows: 64 slots = 2 regs/lane ----
    float tv0[4], tv1[4]; int ti0[4], ti1[4];
    float curMin[4]; int curSlot[4];
#pragma unroll
    for (int r = 0; r < 4; ++r) {
        tv0[r] = tv1[r] = -1.0f; ti0[r] = ti1[r] = 0;
        curMin[r] = -1.0f; curSlot[r] = 0;
    }

    for (int nb = 0; nb < NLAT; nb += NCHUNK) {
        const int ncol0 = nb + wave * 32 + nl;       // B frag 0 column
        const int ncol1 = ncol0 + 16;                // B frag 1 column
        const float benc0 = b_enc[ncol0];
        const float benc1 = b_enc[ncol1];
        const __bf16* wrow0 = wencB + (size_t)ncol0 * DIN;
        const __bf16* wrow1 = wencB + (size_t)ncol1 * DIN;

        // cover post-barrier latency of the next chunk's B rows
        if (nb + NCHUNK < NLAT) {
            __builtin_prefetch(wrow0 + (size_t)NCHUNK * DIN, 0, 3);
            __builtin_prefetch(wrow1 + (size_t)NCHUNK * DIN, 0, 3);
        }

        v8f c00 = {}, c01 = {}, c10 = {}, c11 = {};
#pragma unroll 2
        for (int ks = 0; ks < DIN; ks += 32) {
            union { uint4 u[2]; v16bf v; } b0, b1, a0, a1;
            // B (32x16 bf16): lane = column; lanes 0-15 hold K=0..15,
            // lanes 16-31 hold K=16..31 (contiguous) -> two b128 loads each
            const uint4* bp0 = (const uint4*)(wrow0 + ks + g * 16);
            b0.u[0] = bp0[0]; b0.u[1] = bp0[1];
            const uint4* bp1 = (const uint4*)(wrow1 + ks + g * 16);
            b1.u[0] = bp1[0]; b1.u[1] = bp1[1];
            // A (16x32 bf16): lane = row; K=g*16..g*16+15 (contiguous, matches B)
            const uint4* ap0 = (const uint4*)(&xcb[nl * A_STR + ks + g * 16]);
            a0.u[0] = ap0[0]; a0.u[1] = ap0[1];
            const uint4* ap1 = (const uint4*)(&xcb[(nl + 16) * A_STR + ks + g * 16]);
            a1.u[0] = ap1[0]; a1.u[1] = ap1[1];
            // each A frag feeds 2 WMMAs, each B frag feeds 2 WMMAs
            c00 = __builtin_amdgcn_wmma_f32_16x16x32_bf16(false, a0.v, false, b0.v,
                                                          (short)0, c00, false, false);
            c01 = __builtin_amdgcn_wmma_f32_16x16x32_bf16(false, a0.v, false, b1.v,
                                                          (short)0, c01, false, false);
            c11 = __builtin_amdgcn_wmma_f32_16x16x32_bf16(false, a1.v, false, b1.v,
                                                          (short)0, c11, false, false);
            c10 = __builtin_amdgcn_wmma_f32_16x16x32_bf16(false, a1.v, false, b0.v,
                                                          (short)0, c10, false, false);
        }
        // bias + ReLU, park chunk in LDS (C layout: reg r -> row r+8g, col = lane%16)
        const int colp0 = wave * 32 + nl;
        const int colp1 = colp0 + 16;
#pragma unroll
        for (int r = 0; r < 8; ++r) {
            pre_s[(g * 8 + r) * P_STR + colp0]      = fmaxf(c00[r] + benc0, 0.0f);
            pre_s[(g * 8 + r) * P_STR + colp1]      = fmaxf(c01[r] + benc1, 0.0f);
            pre_s[(16 + g * 8 + r) * P_STR + colp0] = fmaxf(c10[r] + benc0, 0.0f);
            pre_s[(16 + g * 8 + r) * P_STR + colp1] = fmaxf(c11[r] + benc1, 0.0f);
        }
        __syncthreads();

        // ---- top-64 update: wave handles rows wave*4 .. wave*4+3 ----
#pragma unroll 1
        for (int rr = 0; rr < 4; ++rr) {
            const int row = wave * 4 + rr;
#pragma unroll 1
            for (int j = 0; j < 8; ++j) {
                float v = pre_s[row * P_STR + lane + 32 * j];
                unsigned mask = __builtin_amdgcn_ballot_w32(v > curMin[rr]);
                while (mask) {
                    int s = __builtin_ctz(mask); mask &= mask - 1;
                    float cv = bcastf(v, s);
                    int cidx = nb + s + 32 * j;
                    if (cv > curMin[rr]) {
                        if ((curSlot[rr] & 31) == lane) {
                            if (curSlot[rr] < 32) { tv0[rr] = cv; ti0[rr] = cidx; }
                            else                  { tv1[rr] = cv; ti1[rr] = cidx; }
                        }
                        // wave-wide min reduce over 64 slots (tie-break on slot id)
                        float mv; int ms;
                        if (tv0[rr] <= tv1[rr]) { mv = tv0[rr]; ms = lane; }
                        else                    { mv = tv1[rr]; ms = lane + 32; }
#pragma unroll
                        for (int off = 16; off >= 1; off >>= 1) {
                            int xl = (lane ^ off) << 2;
                            float ov = __builtin_bit_cast(float,
                                __builtin_amdgcn_ds_bpermute(xl, __builtin_bit_cast(int, mv)));
                            int os = __builtin_amdgcn_ds_bpermute(xl, ms);
                            if (ov < mv || (ov == mv && os < ms)) { mv = ov; ms = os; }
                        }
                        curMin[rr] = mv; curSlot[rr] = ms;
                    }
                }
            }
        }
        __syncthreads();
    }

    // ---- emit top_acts / top_idx (order irrelevant per reference) ----
    float* out_acts = out + (size_t)BATCH * DIN;
    float* out_idx  = out_acts + (size_t)BATCH * KTOP;
#pragma unroll
    for (int rr = 0; rr < 4; ++rr) {
        const int grow = r0 + wave * 4 + rr;
        out_acts[(size_t)grow * KTOP + lane]      = tv0[rr];
        out_acts[(size_t)grow * KTOP + 32 + lane] = tv1[rr];
        out_idx [(size_t)grow * KTOP + lane]      = (float)ti0[rr];
        out_idx [(size_t)grow * KTOP + 32 + lane] = (float)ti1[rr];
    }

    // ---- sparse decode: sae_out = sum_k act * W_dec[idx] + b_dec ----
#pragma unroll 1
    for (int rr = 0; rr < 4; ++rr) {
        const int grow = r0 + wave * 4 + rr;
        float acc[24];
#pragma unroll
        for (int j = 0; j < 24; ++j) acc[j] = b_dec[lane + 32 * j];
#pragma unroll 1
        for (int s = 0; s < 32; ++s) {
            float a0 = bcastf(tv0[rr], s);
            int   i0 = bcasti(ti0[rr], s);
            if (a0 > 0.0f) {
                const float* wr = w_dec + (size_t)i0 * DIN;
#pragma unroll
                for (int j = 0; j < 24; ++j) acc[j] += a0 * wr[lane + 32 * j];
            }
            float a1 = bcastf(tv1[rr], s);
            int   i1 = bcasti(ti1[rr], s);
            if (a1 > 0.0f) {
                const float* wr = w_dec + (size_t)i1 * DIN;
#pragma unroll
                for (int j = 0; j < 24; ++j) acc[j] += a1 * wr[lane + 32 * j];
            }
        }
#pragma unroll
        for (int j = 0; j < 24; ++j)
            out[(size_t)grow * DIN + lane + 32 * j] = acc[j];
    }
}

extern "C" void kernel_launch(void* const* d_in, const int* in_sizes, int n_in,
                              void* d_out, int out_size, void* d_ws, size_t ws_size,
                              hipStream_t stream) {
    const float* x     = (const float*)d_in[0];
    const float* w_enc = (const float*)d_in[1];
    const float* b_enc = (const float*)d_in[2];
    const float* w_dec = (const float*)d_in[3];
    const float* b_dec = (const float*)d_in[4];
    float* out = (float*)d_out;

    __bf16* wencB = (__bf16*)d_ws;   // 24576*768*2 = 37.75 MB scratch

    const int cvt_blocks = (NLAT * DIN) / (256 * 8);   // 9216
    wenc_to_bf16<<<cvt_blocks, 256, 0, stream>>>(w_enc, wencB);

    sae_fused<<<BATCH / ROWS_WG, 256, 0, stream>>>(x, wencB, b_enc, w_dec, b_dec, out);
}
